// CrossAttentionWithCache_40827959116091
// MI455X (gfx1250) — matrix operand order repaired and checked by reading, not compiled, not run on hardware
//
#include <hip/hip_runtime.h>
#include <hip/hip_bf16.h>
#include <math.h>

// ---------------------------------------------------------------------------
// CrossAttentionWithCache for MI455X (gfx1250, wave32, WMMA).
// Heavy work = K/V projections: [48000,1024] x [1024,1024]^T, twice (201 GF).
// Done with v_wmma_f32_16x16x32_f16 using a 2-term f16 split (3 WMMAs per
// fp32-accurate product). Weights pre-split to f16 hi/lo once (8 MB in d_ws,
// stored contiguously so the GEMM addresses all 4 streams + 4 n-tiles with
// immediate offsets off a single per-lane base pointer).
// ---------------------------------------------------------------------------

typedef __attribute__((ext_vector_type(16))) _Float16 v16h;
typedef __attribute__((ext_vector_type(8)))  float    v8f;

#define DMODEL 1024
#define SLEN   1500
#define BATCH  32
#define NHEAD  16
#define HDIM   64

// ---------------------------------------------------------------------------
// Split a fp32 weight matrix (1024x1024) into f16 hi/lo parts.
// hi = f16(w), lo = f16(w - (float)hi)  -> hi+lo carries ~22 mantissa bits.
// ---------------------------------------------------------------------------
__global__ __launch_bounds__(256) void w_split_kernel(
    const float* __restrict__ W, _Float16* __restrict__ Hi, _Float16* __restrict__ Lo)
{
    int i = (blockIdx.x * 256 + threadIdx.x) * 4;
    float4 w = *(const float4*)(W + i);
    float a[4] = {w.x, w.y, w.z, w.w};
#pragma unroll
    for (int t = 0; t < 4; ++t) {
        _Float16 h = (_Float16)a[t];
        Hi[i + t] = h;
        Lo[i + t] = (_Float16)(a[t] - (float)h);
    }
}

// ---------------------------------------------------------------------------
// Fused K/V projection GEMM.
//   Kout[m][n] = sum_d X[m][d] * Wk[n][d]
//   Vout[m][n] = sum_d X[m][d] * Wv[n][d] + bv[n]
// M = 48000 (=375*128), N = 1024 (=16*64), depth K = 1024.
// Block: 256 thr (8 waves), tile 128M x 64N for BOTH outputs.
// Wave w: rows m0+w*16..+15, 4 WMMA n-tiles x {K,V} -> 8 f32 accumulators.
// Wsplit layout: [4][1024][1024] f16 = { WkHi, WkLo, WvHi, WvLo }.
// ---------------------------------------------------------------------------
__global__ __launch_bounds__(256) void kv_gemm_kernel(
    const float*    __restrict__ X,
    const _Float16* __restrict__ Wsplit,
    const float*    __restrict__ bv,
    float* __restrict__ Kout, float* __restrict__ Vout)
{
    const int lane = threadIdx.x & 31;
    const int wave = threadIdx.x >> 5;
    const int n0   = blockIdx.x * 64;                 // n-blocks fastest -> L2 reuse of X
    const int m0   = blockIdx.y * 128 + wave * 16;

    const int l15  = lane & 15;
    const int hsel = lane >> 4;                       // A: K 0-15 vs 16-31

    // Single advancing pointers; everything else is immediate offsets.
    const float*    xp = X + (size_t)(m0 + l15) * DMODEL + hsel * 16;
    const _Float16* bp = Wsplit + (size_t)(n0 + l15) * DMODEL + hsel * 16;

    constexpr size_t MATS    = (size_t)DMODEL * DMODEL;  // 1M halfs = 2MB
    constexpr size_t JSTRIDE = (size_t)16 * DMODEL;      // 16 rows   = 32KB

    v8f vzero = {};
    v8f accK[4], accV[4];
#pragma unroll
    for (int j = 0; j < 4; ++j) { accK[j] = vzero; accV[j] = vzero; }

    for (int k0 = 0; k0 < DMODEL; k0 += 32, xp += 32, bp += 32) {
        // ---- A operand: contiguous-k A 16x32 lane layout (k = hsel*16 + i), built from fp32 X
        float4 f0 = *(const float4*)(xp);        // K = k0 + hsel*16 + 0..3
        float4 f1 = *(const float4*)(xp + 4);    //                 + 4..7
        float4 f2 = *(const float4*)(xp + 8);    //                 + 8..11
        float4 f3 = *(const float4*)(xp + 12);   //                 + 12..15
        float a[16];
        a[0]=f0.x;  a[1]=f0.y;  a[2]=f0.z;  a[3]=f0.w;
        a[4]=f1.x;  a[5]=f1.y;  a[6]=f1.z;  a[7]=f1.w;
        a[8]=f2.x;  a[9]=f2.y;  a[10]=f2.z; a[11]=f2.w;
        a[12]=f3.x; a[13]=f3.y; a[14]=f3.z; a[15]=f3.w;

        v16h ahi, alo;
#pragma unroll
        for (int i = 0; i < 16; ++i) {
            _Float16 h = (_Float16)a[i];
            ahi[i] = h;
            alo[i] = (_Float16)(a[i] - (float)h);
        }

#pragma unroll
        for (int j = 0; j < 4; ++j) {
            // B operand: lane = output column; lanes 0-15 K 0-15, 16-31 K 16-31.
            const _Float16* bj = bp + (size_t)j * JSTRIDE;
            v16h bkh = *(const v16h*)(bj);
            v16h bkl = *(const v16h*)(bj + MATS);
            v16h bvh = *(const v16h*)(bj + 2 * MATS);
            v16h bvl = *(const v16h*)(bj + 3 * MATS);

            accK[j] = __builtin_amdgcn_wmma_f32_16x16x32_f16(false, ahi, false, bkh,
                                                             (short)0, accK[j], false, false);
            accK[j] = __builtin_amdgcn_wmma_f32_16x16x32_f16(false, ahi, false, bkl,
                                                             (short)0, accK[j], false, false);
            accK[j] = __builtin_amdgcn_wmma_f32_16x16x32_f16(false, alo, false, bkh,
                                                             (short)0, accK[j], false, false);

            accV[j] = __builtin_amdgcn_wmma_f32_16x16x32_f16(false, ahi, false, bvh,
                                                             (short)0, accV[j], false, false);
            accV[j] = __builtin_amdgcn_wmma_f32_16x16x32_f16(false, ahi, false, bvl,
                                                             (short)0, accV[j], false, false);
            accV[j] = __builtin_amdgcn_wmma_f32_16x16x32_f16(false, alo, false, bvh,
                                                             (short)0, accV[j], false, false);
        }
    }

    // Epilogue: C/D layout -> vgpr r = row (m0 + r + hsel*8), lane&15 = col.
    // Row stride 4KB becomes immediate store offsets.
#pragma unroll
    for (int j = 0; j < 4; ++j) {
        const int col = n0 + j * 16 + l15;
        const float bvc = bv[col];
        const size_t base = (size_t)(m0 + hsel * 8) * DMODEL + col;
        float* kp = Kout + base;
        float* vp = Vout + base;
#pragma unroll
        for (int r = 0; r < 8; ++r) {
            kp[(size_t)r * DMODEL] = accK[j][r];
            vp[(size_t)r * DMODEL] = accV[j][r] + bvc;
        }
    }
}

// ---------------------------------------------------------------------------
// Small dense layer: out[b][n] = dot(x[b], W[n]) + bias[n].  x: [32,1024].
// One block per batch row. Tiny (67 MFLOP) -> plain VALU.
// ---------------------------------------------------------------------------
__global__ __launch_bounds__(256) void vec_linear_kernel(
    const float* __restrict__ x, const float* __restrict__ W,
    const float* __restrict__ bias, float* __restrict__ out)
{
    __shared__ float sx[DMODEL];
    const int b = blockIdx.x;
    for (int i = threadIdx.x; i < DMODEL; i += 256) sx[i] = x[b * DMODEL + i];
    __syncthreads();
    for (int n = threadIdx.x; n < DMODEL; n += 256) {
        const float4* wr = (const float4*)(W + (size_t)n * DMODEL);
        float acc = 0.f;
#pragma unroll 4
        for (int k4 = 0; k4 < DMODEL / 4; ++k4) {
            float4 w = wr[k4];
            acc += w.x * sx[4 * k4] + w.y * sx[4 * k4 + 1]
                 + w.z * sx[4 * k4 + 2] + w.w * sx[4 * k4 + 3];
        }
        out[b * DMODEL + n] = acc + bias[n];
    }
}

// ---------------------------------------------------------------------------
// Attention: one block per (head, batch). scores = (q . k) * 1/8 (SCALE^2),
// fp32 softmax over S=1500, then ctx = P @ V. Memory-bound (re-reads K,V).
// ---------------------------------------------------------------------------
__global__ __launch_bounds__(256) void attn_kernel(
    const float* __restrict__ q,     // [32,1024]
    const float* __restrict__ K,     // [32,1500,1024]
    const float* __restrict__ V,     // [32,1500,1024]
    float* __restrict__ ctx)         // [32,1024]
{
    __shared__ float sc[SLEN];
    __shared__ float red[256];
    __shared__ float sq[HDIM];

    const int h = blockIdx.x, b = blockIdx.y, tid = threadIdx.x;
    const size_t baseBH = (size_t)b * SLEN * DMODEL + (size_t)h * HDIM;

    if (tid < HDIM) sq[tid] = q[b * DMODEL + h * HDIM + tid];
    __syncthreads();

    // scores + running max
    float lmax = -3.4e38f;
    for (int s = tid; s < SLEN; s += 256) {
        const float4* kr = (const float4*)(K + baseBH + (size_t)s * DMODEL);
        float acc = 0.f;
#pragma unroll
        for (int i = 0; i < HDIM / 4; ++i) {
            float4 kv = kr[i];
            acc += kv.x * sq[4 * i] + kv.y * sq[4 * i + 1]
                 + kv.z * sq[4 * i + 2] + kv.w * sq[4 * i + 3];
        }
        acc *= 0.125f;                 // SCALE^2 = HEAD_DIM^-0.5 = 1/8
        sc[s] = acc;
        lmax = fmaxf(lmax, acc);
    }
    red[tid] = lmax; __syncthreads();
    for (int off = 128; off > 0; off >>= 1) {
        if (tid < off) red[tid] = fmaxf(red[tid], red[tid + off]);
        __syncthreads();
    }
    const float gmax = red[0];
    __syncthreads();

    // exp + sum
    float lsum = 0.f;
    for (int s = tid; s < SLEN; s += 256) {
        float e = expf(sc[s] - gmax);
        sc[s] = e;
        lsum += e;
    }
    red[tid] = lsum; __syncthreads();
    for (int off = 128; off > 0; off >>= 1) {
        if (tid < off) red[tid] += red[tid + off];
        __syncthreads();
    }
    const float inv = 1.0f / red[0];
    __syncthreads();

    // ctx = P @ V : thread (c,d), 4 s-chunks x 64 dims (coalesced V reads)
    const int d = tid & 63, c = tid >> 6;
    float acc = 0.f;
    for (int s = c; s < SLEN; s += 4)
        acc += sc[s] * V[baseBH + (size_t)s * DMODEL + d];
    red[tid] = acc; __syncthreads();
    if (tid < HDIM)
        ctx[b * DMODEL + h * HDIM + tid] =
            (red[tid] + red[64 + tid] + red[128 + tid] + red[192 + tid]) * inv;
}

// ---------------------------------------------------------------------------
extern "C" void kernel_launch(void* const* d_in, const int* in_sizes, int n_in,
                              void* d_out, int out_size, void* d_ws, size_t ws_size,
                              hipStream_t stream)
{
    (void)in_sizes; (void)n_in; (void)out_size; (void)ws_size;

    const float* hidden = (const float*)d_in[0];   // [32,1,1024]
    const float* enc    = (const float*)d_in[1];   // [32,1500,1024]
    const float* Wq     = (const float*)d_in[2];
    const float* bq     = (const float*)d_in[3];
    const float* Wk     = (const float*)d_in[4];
    const float* Wv     = (const float*)d_in[5];
    const float* bv     = (const float*)d_in[6];
    const float* Wo     = (const float*)d_in[7];
    const float* bo     = (const float*)d_in[8];

    float* out      = (float*)d_out;
    float* attn_out = out;                                    // [32,1024]
    float* Kout     = out + (size_t)BATCH * DMODEL;           // [32,1500,1024]
    float* Vout     = Kout + (size_t)BATCH * SLEN * DMODEL;   // [32,1500,1024]

    // workspace: contiguous [4][1024][1024] f16 split weights + q + ctx (~8.4 MB)
    _Float16* wsplit = (_Float16*)d_ws;                       // {kHi,kLo,vHi,vLo}
    _Float16* wk_hi  = wsplit;
    _Float16* wk_lo  = wk_hi + (size_t)DMODEL * DMODEL;
    _Float16* wv_hi  = wk_lo + (size_t)DMODEL * DMODEL;
    _Float16* wv_lo  = wv_hi + (size_t)DMODEL * DMODEL;
    float*    q_ws   = (float*)(wv_lo + (size_t)DMODEL * DMODEL);
    float*    c_ws   = q_ws + (size_t)BATCH * DMODEL;

    w_split_kernel<<<(DMODEL * DMODEL) / (256 * 4), 256, 0, stream>>>(Wk, wk_hi, wk_lo);
    w_split_kernel<<<(DMODEL * DMODEL) / (256 * 4), 256, 0, stream>>>(Wv, wv_hi, wv_lo);

    vec_linear_kernel<<<BATCH, 256, 0, stream>>>(hidden, Wq, bq, q_ws);

    // M=48000 -> 375 row-blocks of 128; N=1024 -> 16 col-blocks of 64 (fastest)
    kv_gemm_kernel<<<dim3(16, 375), 256, 0, stream>>>(enc, wsplit, bv, Kout, Vout);

    attn_kernel<<<dim3(NHEAD, BATCH), 256, 0, stream>>>(q_ws, Kout, Vout, c_ws);

    vec_linear_kernel<<<BATCH, 256, 0, stream>>>(c_ws, Wo, bo, attn_out);
}
